// WaveAttention_11802570129598
// MI455X (gfx1250) — matrix operand order repaired and checked by reading, not compiled, not run on hardware
//
#include <hip/hip_runtime.h>
#include <hip/hip_bf16.h>
#include <stdint.h>

typedef _Float16 f16;
typedef __attribute__((ext_vector_type(16))) _Float16 v16h;
typedef __attribute__((ext_vector_type(8)))  _Float16 v8h;
typedef __attribute__((ext_vector_type(8)))  float    v8f;

// ---------------------------------------------------------------------------
// Kernel 1: fold Haar butterfly into W1 -> Wf[o][pix*256+c] (f16), W2 -> f16.
// ---------------------------------------------------------------------------
__global__ __launch_bounds__(256) void prep_weights(
    const float* __restrict__ W1,   // [256][1024]  (o, s*256+c), s = LL,LH,HL,HH
    const float* __restrict__ W2,   // [256][256]
    f16* __restrict__ Wf,           // [256][1024]
    f16* __restrict__ W2h)          // [256][256]
{
    const int o = blockIdx.x;
    const int c = threadIdx.x;
    const float* w = W1 + (size_t)o * 1024;
    const float s0 = w[c], s1 = w[256 + c], s2 = w[512 + c], s3 = w[768 + c];
    f16* wo = Wf + (size_t)o * 1024;
    wo[0 * 256 + c] = (f16)(0.5f * (s0 + s1 + s2 + s3));   // pixel (0,0)
    wo[1 * 256 + c] = (f16)(0.5f * (s0 + s1 - s2 - s3));   // pixel (0,1)
    wo[2 * 256 + c] = (f16)(0.5f * (s0 - s1 + s2 - s3));   // pixel (1,0)
    wo[3 * 256 + c] = (f16)(0.5f * (s0 - s1 - s2 + s3));   // pixel (1,1)
    W2h[(size_t)o * 256 + c] = (f16)W2[(size_t)o * 256 + c];
}

// ---------------------------------------------------------------------------
// Async staging: ROWS x 32-half B tile moved global->LDS with
// GLOBAL_LOAD_ASYNC_TO_LDS_B128 (no VGPR data path, ASYNCcnt-tracked).
// 16B chunks: ROWS*4 total, ROWS/32 per thread.  LDS row stride 40 halves
// (80 B) -> fragment reads map 16 lanes onto distinct bank groups.
// NOTE: ISA adds INST_OFFSET to both LDS and global addresses, so each chunk
// carries its own computed operands (no immediate-offset reuse).
// ---------------------------------------------------------------------------
template<int CHUNKS, int ROWSTRIDE>
__device__ __forceinline__ void async_stage(const f16* __restrict__ src,
                                            int kt, uint32_t ldsBase, int t) {
    #pragma unroll
    for (int c = 0; c < CHUNKS; ++c) {
        const int lin = c * 128 + t;
        const int row = lin >> 2;
        const int col = lin & 3;
        const unsigned long long g =
            (unsigned long long)(uintptr_t)(src + (size_t)row * ROWSTRIDE
                                                + kt + col * 8);
        const uint32_t l = ldsBase + (uint32_t)(row * 80 + col * 16);
        asm volatile("global_load_async_to_lds_b128 %0, %1, off"
                     :: "v"(l), "v"(g) : "memory");
    }
}
__device__ __forceinline__ void wait_async() {
    asm volatile("s_wait_asynccnt 0x0" ::: "memory");
}

// ---------------------------------------------------------------------------
// Kernel 2: fused  y2 = (relu(A*WfT + b1)) * W2T + b2
// Block = 64 M rows (wave32 = 16 rows), N = 256.  B tiles streamed into LDS
// with async-to-LDS double buffering; consumption is ds_load fragments ->
// v_wmma_f32_16x16x32_f16 (f32 accumulate).
// ---------------------------------------------------------------------------
__global__ __launch_bounds__(128) void fused_gemm(
    const float* __restrict__ x,    // [8][16384][256]
    const f16*  __restrict__ Wf,    // [256][1024]
    const float* __restrict__ b1,   // [256]
    const f16*  __restrict__ W2h,   // [256][256]
    const float* __restrict__ b2,   // [256]
    float* __restrict__ y2)         // [32768][256]  (m = b*4096 + h'*64 + w')
{
    // LDS arena:
    //   phase 1: Bst = [2][256][40] f16  @ 0      (40960 B)
    //   phase 2: Y1  = [4][16][264] f16  @ 0      (33792 B, reuses Bst)
    //            W2s = [2][128][40] f16  @ 33792  (20480 B) -> total 54272
    __shared__ __align__(128) char smem[54784];
    f16* const Bst = (f16*)smem;
    f16* const Y1  = (f16*)smem;
    f16* const W2s = (f16*)(smem + 33792);
    const uint32_t ldsBase = (uint32_t)(uintptr_t)(void*)smem;  // LDS byte offset

    const int t      = threadIdx.x;
    const int lane   = t & 31;
    const int wave   = t >> 5;
    const int m0     = blockIdx.x * 64 + wave * 16;  // this wave's M base
    const int bImg   = m0 >> 12;
    const int rem    = m0 & 4095;
    const int hp     = rem >> 6;                     // h'
    const int w0     = rem & 63;                     // w' base
    const int mrow   = lane & 15;                    // A row / B column
    const int hiHalf = lane >> 4;

    // A row = DWT position (bImg, hp, w0+mrow): 4 source pixels in x.
    const long nA = (long)bImg * 16384 + (long)(2 * hp) * 128 + 2 * (w0 + mrow);
    long xbase[4];
    xbase[0] = (nA + 0)   * 256;
    xbase[1] = (nA + 1)   * 256;
    xbase[2] = (nA + 128) * 256;
    xbase[3] = (nA + 129) * 256;

    const v8f vzero = {};
    v8f c1[16];
    #pragma unroll
    for (int nt = 0; nt < 16; ++nt) c1[nt] = vzero;

    // ---- GEMM1: K = 1024, 32-wide steps, async double-buffered B tiles ----
    async_stage<8, 1024>(Wf, 0, ldsBase, t);         // prologue -> buf0
    for (int ks = 0; ks < 32; ++ks) {
        const int kt = ks * 32;
        f16* const buf = Bst + (ks & 1) * (256 * 40);
        wait_async();                                // my tile writes done
        __syncthreads();                             // everyone's done; also
                                                     // guards buffer overwrite
        if (ks < 31)                                 // next step in flight
            async_stage<8, 1024>(Wf, kt + 32,
                                 ldsBase + ((ks + 1) & 1) * 20480, t);

        const int pix = kt >> 8;
        const int c0  = (kt & 255) + hiHalf * 16;
        const float* pa = x + xbase[pix] + c0;
        const v8f f0 = *(const v8f*)(pa);            // K-local {0..7}/{16..23}
        const v8f f1 = *(const v8f*)(pa + 8);        // K-local {8..15}/{24..31}
        v16h a;
        #pragma unroll
        for (int i = 0; i < 8; ++i) { a[i] = (f16)f0[i]; a[i + 8] = (f16)f1[i]; }

        #pragma unroll
        for (int nt = 0; nt < 16; ++nt) {
            const v16h bf = *(const v16h*)(buf + (size_t)(nt * 16 + mrow) * 40
                                               + 16 * hiHalf);
            c1[nt] = __builtin_amdgcn_wmma_f32_16x16x32_f16(
                false, a, false, bf, (short)0, c1[nt], false, false);
        }
    }
    __syncthreads();   // all waves done reading Bst before Y1 overwrites it

    // ---- epilogue 1: bias + relu + f16 -> Y1 (row-major [m_local][o]) ----
    f16* const y1w = Y1 + (size_t)wave * (16 * 264);
    #pragma unroll
    for (int nt = 0; nt < 16; ++nt) {
        const int o = nt * 16 + mrow;                // C-layout: N = lane%16
        const float bias = b1[o];
        #pragma unroll
        for (int r = 0; r < 8; ++r) {                // C-layout: M = r+8*hiHalf
            float v = c1[nt][r] + bias;
            v = v > 0.0f ? v : 0.0f;
            y1w[(size_t)(r + 8 * hiHalf) * 264 + o] = (f16)v;
        }
    }
    __syncthreads();

    // ---- GEMM2: K = 256, 16 pipeline stages (8 K-steps x 2 N-halves),
    //      async double-buffered [128][40] W2 tiles ----
    v8f c2[16];
    #pragma unroll
    for (int nt = 0; nt < 16; ++nt) c2[nt] = vzero;

    async_stage<4, 256>(W2h, 0, ldsBase + 33792, t); // stage 0 (ks=0, nh=0)
    for (int s = 0; s < 16; ++s) {
        const int ks = s >> 1;
        const int nh = s & 1;
        const int kt = ks * 32;
        wait_async();
        __syncthreads();
        if (s < 15) {
            const int s2  = s + 1;
            const int ks2 = s2 >> 1;
            const int nh2 = s2 & 1;
            async_stage<4, 256>(W2h + (size_t)(nh2 * 128) * 256, ks2 * 32,
                                ldsBase + 33792 + (s2 & 1) * 10240, t);
        }

        const f16* la = y1w + (size_t)mrow * 264 + kt + 16 * hiHalf;
        const v8h lo = *(const v8h*)(la);
        const v8h hi = *(const v8h*)(la + 8);
        const v16h a2 = __builtin_shufflevector(lo, hi,
            0, 1, 2, 3, 4, 5, 6, 7, 8, 9, 10, 11, 12, 13, 14, 15);

        f16* const wbuf = W2s + (s & 1) * (128 * 40);
        #pragma unroll
        for (int j = 0; j < 8; ++j) {
            const int nt = nh * 8 + j;               // B row o = nt*16+mrow
            const v16h bf = *(const v16h*)(wbuf + (size_t)(j * 16 + mrow) * 40
                                                + 16 * hiHalf);
            c2[nt] = __builtin_amdgcn_wmma_f32_16x16x32_f16(
                false, a2, false, bf, (short)0, c2[nt], false, false);
        }
    }

    // ---- epilogue 2: bias + store y2[m][o] (f32, coalesced) ----
    #pragma unroll
    for (int nt = 0; nt < 16; ++nt) {
        const int o = nt * 16 + mrow;
        const float bias = b2[o];
        #pragma unroll
        for (int r = 0; r < 8; ++r) {
            const long m = m0 + r + 8 * hiHalf;
            y2[m * 256 + o] = c2[nt][r] + bias;
        }
    }
}

// ---------------------------------------------------------------------------
// Kernel 3: bilinear 64->128 (half-pixel, clamped/normalized edges) + layout
// change to out[b][h*128+w][o].  Pure bandwidth pass, channel-coalesced.
// ---------------------------------------------------------------------------
__global__ __launch_bounds__(256) void resize_out(
    const float* __restrict__ y2,   // [8][64][64][256]
    float* __restrict__ out)        // [8][128*128][256]
{
    const long gid = (long)blockIdx.x * 256 + threadIdx.x;
    const int  o   = (int)(gid & 255);
    const long pix = gid >> 8;                 // b*16384 + h*128 + w
    const int  w   = (int)(pix & 127);
    const int  h   = (int)((pix >> 7) & 127);
    const long b   = pix >> 14;

    const int jh = h >> 1;
    int h0, h1; float wh0;
    if (h & 1) { h0 = jh; h1 = (jh + 1 < 64) ? jh + 1 : 63; wh0 = 0.75f; }
    else       { h0 = (jh - 1 >= 0) ? jh - 1 : 0; h1 = jh;  wh0 = 0.25f; }
    const float wh1 = 1.0f - wh0;

    const int jw = w >> 1;
    int w0i, w1i; float ww0;
    if (w & 1) { w0i = jw; w1i = (jw + 1 < 64) ? jw + 1 : 63; ww0 = 0.75f; }
    else       { w0i = (jw - 1 >= 0) ? jw - 1 : 0; w1i = jw;  ww0 = 0.25f; }
    const float ww1 = 1.0f - ww0;

    const long base = b * 4096;
    const float v00 = y2[(base + (long)h0 * 64 + w0i) * 256 + o];
    const float v01 = y2[(base + (long)h0 * 64 + w1i) * 256 + o];
    const float v10 = y2[(base + (long)h1 * 64 + w0i) * 256 + o];
    const float v11 = y2[(base + (long)h1 * 64 + w1i) * 256 + o];

    out[gid] = wh0 * (ww0 * v00 + ww1 * v01) + wh1 * (ww0 * v10 + ww1 * v11);
}

// ---------------------------------------------------------------------------
extern "C" void kernel_launch(void* const* d_in, const int* in_sizes, int n_in,
                              void* d_out, int out_size, void* d_ws, size_t ws_size,
                              hipStream_t stream) {
    const float* x  = (const float*)d_in[0];   // [8][16384][256]
    const float* W1 = (const float*)d_in[1];   // [256][1024]
    const float* b1 = (const float*)d_in[2];   // [256]
    const float* W2 = (const float*)d_in[3];   // [256][256]
    const float* b2 = (const float*)d_in[4];   // [256]
    float* out = (float*)d_out;                // [8][16384][256]

    char* ws = (char*)d_ws;
    f16*   Wf  = (f16*)(ws);                        // 512 KiB
    f16*   W2h = (f16*)(ws + 524288);               // 128 KiB
    float* y2  = (float*)(ws + 524288 + 131072);    // 32 MiB

    prep_weights<<<256, 256, 0, stream>>>(W1, W2, Wf, W2h);
    fused_gemm<<<512, 128, 0, stream>>>(x, Wf, b1, W2h, b2, y2);
    resize_out<<<131072, 256, 0, stream>>>(y2, out);
}
